// RNN_1090921693291
// MI455X (gfx1250) — compile-verified, matrix operand order checked by reading, not run on hardware
//
#include <hip/hip_runtime.h>
#include <hip/hip_bf16.h>
#include <stdint.h>

typedef __attribute__((ext_vector_type(16))) _Float16 v16h;
typedef __attribute__((ext_vector_type(8)))  float    v8f;

#define T_STEPS 400
#define BATCH   32
#define TB      (T_STEPS * BATCH)   // 12800
#define D_IN    440
#define D_PAD   448
#define H_DIM   1024
#define C_OUT   3500
#define C_PAD   3584

// ---------------------------------------------------------------------------
// Tensor Data Mover support (CDNA5). Guarded so either toolchain compiles.
// This toolchain (clang-23 / therock-10.0 headers) exposes the 6-arg builtin:
//   (uint32x4 g0, int32x8 g1, int32x4 g2, int32x4 g3, int32x8 g4, i32 cpol)
// ---------------------------------------------------------------------------
#if defined(__HIP_DEVICE_COMPILE__) && \
    __has_builtin(__builtin_amdgcn_tensor_load_to_lds) && \
    __has_builtin(__builtin_amdgcn_s_wait_tensorcnt)
#define HAVE_TDM 1
#else
#define HAVE_TDM 0
#endif

#if HAVE_TDM
typedef __attribute__((ext_vector_type(4))) unsigned tdm_v4u;
typedef __attribute__((ext_vector_type(8))) int      tdm_v8i;
typedef __attribute__((ext_vector_type(4))) int      tdm_v4i;

// 2-D tile DMA: global (row-major, strideElems between rows, 2-byte elements)
// -> LDS, rows stored contiguously [tileH][tileW]. Tile always in-bounds.
__device__ __forceinline__ void tdm_load_2d(const void* gptr, void* ldsPtr,
                                            unsigned tileW, unsigned tileH,
                                            unsigned strideElems) {
    unsigned long long ga = (unsigned long long)(uintptr_t)gptr;
    unsigned lds = (unsigned)(uintptr_t)ldsPtr;
    tdm_v4u g0;
    g0[0] = 1u;                                                 // count=1 (user D#)
    g0[1] = lds;                                                // lds_addr
    g0[2] = (unsigned)(ga & 0xFFFFFFFFu);                       // global_addr[31:0]
    g0[3] = (unsigned)((ga >> 32) & 0x01FFFFFFu) | (2u << 30);  // addr[56:32] | type=2
    tdm_v8i g1;
    g1[0] = (int)(1u << 16);                                    // wg_mask=0, data_size=1 (2B)
    g1[1] = (int)((tileW & 0xFFFFu) << 16);                     // tensor_dim0[15:0]
    g1[2] = (int)(((tileW >> 16) & 0xFFFFu) | ((tileH & 0xFFFFu) << 16)); // dim0 hi | dim1 lo
    g1[3] = (int)(((tileH >> 16) & 0xFFFFu) | ((tileW & 0xFFFFu) << 16)); // dim1 hi | tile_dim0
    g1[4] = (int)(tileH & 0xFFFFu);                             // tile_dim1 (tile_dim2=0)
    g1[5] = (int)strideElems;                                   // tensor_dim0_stride[31:0]
    g1[6] = 0;
    g1[7] = 0;
    tdm_v4i gz4 = (tdm_v4i){0, 0, 0, 0};
    tdm_v8i gz8 = (tdm_v8i){0, 0, 0, 0, 0, 0, 0, 0};
    __builtin_amdgcn_tensor_load_to_lds(g0, g1, gz4, gz4, gz8, 0);
}
#endif

// ---------------------------------------------------------------------------
// fp32 -> f16 convert with zero padding (rows and K)
// ---------------------------------------------------------------------------
__global__ void cvt_pad_f16(const float* __restrict__ src, _Float16* __restrict__ dst,
                            int rows, int K, int rowsPad, int Kpad) {
    long i = (long)blockIdx.x * blockDim.x + threadIdx.x;
    long total = (long)rowsPad * Kpad;
    if (i >= total) return;
    int r = (int)(i / Kpad), k = (int)(i % Kpad);
    float v = (r < rows && k < K) ? src[(long)r * K + k] : 0.0f;
    dst[i] = (_Float16)v;
}

__global__ void zero2(float* o) { if (threadIdx.x < 2) o[threadIdx.x] = 0.0f; }

// ---------------------------------------------------------------------------
// WMMA f16 GEMM: C[M,outN] = A[M,Kp] * W[N,Kp]^T + bias, 128x128 tiles.
// grid = (Npad/128, M/128), block = 256 (8 waves). Wave w owns rows w*16..+16.
// Tiles staged global->LDS by the Tensor Data Mover, double buffered.
// ---------------------------------------------------------------------------
__global__ __launch_bounds__(256) void gemm_wmma_f16(
    const _Float16* __restrict__ A, const _Float16* __restrict__ W,
    const float* __restrict__ bias, float* __restrict__ Cout,
    int Kp, int outN, int nLimit) {
    __shared__ __align__(16) _Float16 lA[2][128 * 32];
    __shared__ __align__(16) _Float16 lB[2][128 * 32];
    const int tid  = threadIdx.x;
    const int wave = tid >> 5, lane = tid & 31;
    const int mBase = blockIdx.y * 128;
    const int nBase = blockIdx.x * 128;
    const int laneLo = lane & 15;
    const int koff   = (lane >> 4) << 3;   // +8 for upper half-lanes

    v8f acc[8];
#pragma unroll
    for (int j = 0; j < 8; ++j) acc[j] = (v8f){0.f,0.f,0.f,0.f,0.f,0.f,0.f,0.f};

    const int nIter = Kp >> 5;
    const _Float16* Abase = A + (size_t)mBase * Kp;
    const _Float16* Wbase = W + (size_t)nBase * Kp;

#if HAVE_TDM
    if (wave == 0) {
        tdm_load_2d(Abase, &lA[0][0], 32, 128, (unsigned)Kp);
        tdm_load_2d(Wbase, &lB[0][0], 32, 128, (unsigned)Kp);
    }
#endif
    for (int i = 0; i < nIter; ++i) {
        const int k0 = i << 5;
        const int buf = i & 1;
#if HAVE_TDM
        if (i) __syncthreads();            // buf[(i+1)&1] readers (iter i-1) done
        if (wave == 0) {
            if (i + 1 < nIter) {
                tdm_load_2d(Abase + k0 + 32, &lA[buf ^ 1][0], 32, 128, (unsigned)Kp);
                tdm_load_2d(Wbase + k0 + 32, &lB[buf ^ 1][0], 32, 128, (unsigned)Kp);
                __builtin_amdgcn_s_wait_tensorcnt(2);   // pair i landed
            } else {
                __builtin_amdgcn_s_wait_tensorcnt(0);
            }
        }
        __syncthreads();
#else
        // manual staging fallback: 128 rows x 32 halves = 512 x 16B chunks each
        for (int c = tid; c < 512; c += 256) {
            int row = c >> 2, ch = c & 3;
            const _Float16* ga = Abase + (size_t)row * Kp + k0 + ch * 8;
            ((uint4*)&lA[0][0])[c] = *(const uint4*)ga;
            if (k0 + 32 < Kp) __builtin_prefetch(ga + 32, 0, 1);
        }
        for (int c = tid; c < 512; c += 256) {
            int row = c >> 2, ch = c & 3;
            const _Float16* gb = Wbase + (size_t)row * Kp + k0 + ch * 8;
            ((uint4*)&lB[0][0])[c] = *(const uint4*)gb;
            if (k0 + 32 < Kp) __builtin_prefetch(gb + 32, 0, 1);
        }
        __syncthreads();
#endif
        const _Float16* tA = HAVE_TDM ? &lA[buf][0] : &lA[0][0];
        const _Float16* tB = HAVE_TDM ? &lB[buf][0] : &lB[0][0];

        // A fragment for this wave's 16 rows
        const int mrow = wave * 16 + laneLo;
        union { v16h v; unsigned u[8]; } af;
#pragma unroll
        for (int p = 0; p < 8; ++p) {
            int kk = ((p & 4) ? 16 : 0) + koff + ((p & 3) << 1);
            af.u[p] = *(const unsigned*)&tA[mrow * 32 + kk];
        }
#pragma unroll
        for (int j = 0; j < 8; ++j) {
            int nrow = j * 16 + laneLo;
            union { v16h v; unsigned u[8]; } bf;
#pragma unroll
            for (int p = 0; p < 8; ++p) {
                int kk = ((p & 4) ? 16 : 0) + koff + ((p & 3) << 1);
                bf.u[p] = *(const unsigned*)&tB[nrow * 32 + kk];
            }
            acc[j] = __builtin_amdgcn_wmma_f32_16x16x32_f16(
                false, af.v, false, bf.v, (short)0, acc[j], false, false);
        }
#if !HAVE_TDM
        __syncthreads();
#endif
    }

    // epilogue: row = base + r + 8*(lane>=16), col = nBase + j*16 + lane%16
    const int rowBase = mBase + wave * 16 + koff;
#pragma unroll
    for (int j = 0; j < 8; ++j) {
        int col = nBase + j * 16 + laneLo;
        if (col < nLimit) {
            float bb = bias ? bias[col] : 0.0f;
#pragma unroll
            for (int r = 0; r < 8; ++r)
                Cout[(size_t)(rowBase + r) * outN + col] = acc[j][r] + bb;
        }
    }
}

// ---------------------------------------------------------------------------
// Recurrent layer: one persistent workgroup (1024 thr = 32 waves) loops over T.
// LDS: h (32x1024 f16, 64KB) + sums (256B) + per-wave Uh slice double buffers
// (2 x 32 waves x 32x32 f16 = 128KB). Wave w owns columns [w*32, w*32+32):
// 2 M-tiles x 2 N-tiles. Each wave TDM-streams its private Uh slice, so the
// WMMA K-loop needs no cross-wave barriers.
// ---------------------------------------------------------------------------
#define REC_LDS_BYTES (65536 + 256 + 2 * 65536)

__global__ __launch_bounds__(1024) void rnn_recurrence(
    const float* __restrict__ wx_out,  // [T*32*1024]
    float* __restrict__ pre,           // in/out skip buffer [T*32*1024]
    const _Float16* __restrict__ uh,   // [1024*1024] f16
    const float* __restrict__ gvec, const float* __restrict__ bvec,
    _Float16* __restrict__ hout,       // [T*32*1024] f16
    int isLayer0) {
    extern __shared__ char smem[];
    _Float16* hl     = (_Float16*)smem;                 // 32*1024 f16
    float*    rowsum = (float*)(smem + 65536);          // [32]+[32]
    char*     ubase  = smem + 65536 + 256;              // 2 x 64KB slice buffers

    const int tid  = threadIdx.x;
    const int wave = tid >> 5, lane = tid & 31;
    const int laneLo = lane & 15;
    const int koff   = (lane >> 4) << 3;     // 0 or 8
    const int cbase  = wave * 32;            // this wave's first Uh row / h col

    _Float16* slice[2];
    slice[0] = (_Float16*)(ubase) + wave * (32 * 32);
    slice[1] = (_Float16*)(ubase + 65536) + wave * (32 * 32);
    const _Float16* uwave = uh + (size_t)cbase * H_DIM;

    // zero h_lds (16384 dwords)
    for (int i = tid; i < 32 * 1024 / 2; i += 1024) ((unsigned*)hl)[i] = 0u;
    // per-lane layernorm params (2 columns)
    float gj[2], bj[2];
#pragma unroll
    for (int j = 0; j < 2; ++j) {
        int c = cbase + j * 16 + laneLo;
        gj[j] = gvec[c]; bj[j] = bvec[c];
    }
    __syncthreads();

    for (int t = 0; t < T_STEPS; ++t) {
        const size_t base = (size_t)t * BATCH * H_DIM;
        v8f acc[2][2];
        // init accumulators with wx_out[t] in fragment layout
#pragma unroll
        for (int m = 0; m < 2; ++m)
#pragma unroll
            for (int j = 0; j < 2; ++j) {
                int c = cbase + j * 16 + laneLo;
#pragma unroll
                for (int r = 0; r < 8; ++r)
                    acc[m][j][r] = wx_out[base + (size_t)(m * 16 + koff + r) * H_DIM + c];
            }

#if HAVE_TDM
        tdm_load_2d(uwave, slice[0], 32, 32, H_DIM);    // per-wave DMA, k0 = 0
#endif
        // at += h_{t-1} @ Uh^T
        for (int i = 0; i < H_DIM / 32; ++i) {
            const int k0 = i << 5;
#if HAVE_TDM
            if (i + 1 < H_DIM / 32) {
                tdm_load_2d(uwave + k0 + 32, slice[(i + 1) & 1], 32, 32, H_DIM);
                __builtin_amdgcn_s_wait_tensorcnt(1);   // slice i landed
            } else {
                __builtin_amdgcn_s_wait_tensorcnt(0);
            }
            const _Float16* us = slice[i & 1];
#endif
            union { v16h v; unsigned u[8]; } af[2];
#pragma unroll
            for (int m = 0; m < 2; ++m) {
                int arow = m * 16 + laneLo;
#pragma unroll
                for (int p = 0; p < 8; ++p) {
                    int kk = ((p & 4) ? 16 : 0) + koff + ((p & 3) << 1);
                    af[m].u[p] = *(const unsigned*)&hl[arow * H_DIM + k0 + kk];
                }
            }
#pragma unroll
            for (int j = 0; j < 2; ++j) {
                union { v16h v; unsigned u[8]; } bf;
#pragma unroll
                for (int p = 0; p < 8; ++p) {
                    int kk = ((p & 4) ? 16 : 0) + koff + ((p & 3) << 1);
#if HAVE_TDM
                    bf.u[p] = *(const unsigned*)&us[(j * 16 + laneLo) * 32 + kk];
#else
                    bf.u[p] = *(const unsigned*)&uh[(size_t)(cbase + j * 16 + laneLo) * H_DIM + k0 + kk];
#endif
                }
#pragma unroll
                for (int m = 0; m < 2; ++m)
                    acc[m][j] = __builtin_amdgcn_wmma_f32_16x16x32_f16(
                        false, af[m].v, false, bf.v, (short)0, acc[m][j], false, false);
            }
        }
        __syncthreads();                       // all waves done reading hl / prev sums
        if (tid < 64) rowsum[tid] = 0.0f;      // zero sum + sumsq arrays

        // pre-act store, skip-subtract (in-place pre buffer: read before write)
#pragma unroll
        for (int m = 0; m < 2; ++m)
#pragma unroll
            for (int j = 0; j < 2; ++j) {
                int c = cbase + j * 16 + laneLo;
#pragma unroll
                for (int r = 0; r < 8; ++r) {
                    size_t idx = base + (size_t)(m * 16 + koff + r) * H_DIM + c;
                    float p = acc[m][j][r];
                    float prev = isLayer0 ? 0.0f : pre[idx];
                    pre[idx] = p;
                    acc[m][j][r] = p - prev;   // "a" for layernorm
                }
            }
        __syncthreads();

        // row reductions: shuffle over 16-lane groups (covers wave's 32 cols),
        // then LDS atomics across waves
#pragma unroll
        for (int m = 0; m < 2; ++m)
#pragma unroll
            for (int r = 0; r < 8; ++r) {
                float s = acc[m][0][r] + acc[m][1][r];
                float q = acc[m][0][r] * acc[m][0][r] + acc[m][1][r] * acc[m][1][r];
                for (int msk = 8; msk >= 1; msk >>= 1) {
                    s += __shfl_xor(s, msk, 16);
                    q += __shfl_xor(q, msk, 16);
                }
                if (laneLo == 0) {
                    int row = m * 16 + koff + r;
                    atomicAdd(&rowsum[row], s);
                    atomicAdd(&rowsum[32 + row], q);
                }
            }
        __syncthreads();

        // layernorm (std ddof=1, eps added to std) + relu; write h to LDS+global
#pragma unroll
        for (int m = 0; m < 2; ++m)
#pragma unroll
            for (int r = 0; r < 8; ++r) {
                int row = m * 16 + koff + r;
                float sum = rowsum[row], sq = rowsum[32 + row];
                float mean = sum * (1.0f / 1024.0f);
                float var  = (sq - sum * sum * (1.0f / 1024.0f)) * (1.0f / 1023.0f);
                var = fmaxf(var, 0.0f);
                float inv = 1.0f / (__builtin_sqrtf(var) + 1e-6f);
#pragma unroll
                for (int j = 0; j < 2; ++j) {
                    int c = cbase + j * 16 + laneLo;
                    float val = gj[j] * (acc[m][j][r] - mean) * inv + bj[j];
                    _Float16 hh = (_Float16)fmaxf(val, 0.0f);
                    hl[row * H_DIM + c] = hh;
                    hout[base + (size_t)row * H_DIM + c] = hh;
                }
            }
        __syncthreads();                       // hl ready for step t+1
    }
}

// ---------------------------------------------------------------------------
// in-place log_softmax over rows of [rows, C]
// ---------------------------------------------------------------------------
__global__ __launch_bounds__(256) void log_softmax_inplace(float* __restrict__ pout, int C) {
    float* p = pout + (size_t)blockIdx.x * C;
    __shared__ float red[8];
    const int tid = threadIdx.x, w = tid >> 5;
    float mx = -__builtin_inff();
    for (int c = tid; c < C; c += 256) mx = fmaxf(mx, p[c]);
    for (int m = 16; m >= 1; m >>= 1) mx = fmaxf(mx, __shfl_xor(mx, m, 32));
    if ((tid & 31) == 0) red[w] = mx;
    __syncthreads();
    float rowmax = red[0];
#pragma unroll
    for (int i = 1; i < 8; ++i) rowmax = fmaxf(rowmax, red[i]);
    __syncthreads();
    float s = 0.0f;
    for (int c = tid; c < C; c += 256) s += __expf(p[c] - rowmax);
    for (int m = 16; m >= 1; m >>= 1) s += __shfl_xor(s, m, 32);
    if ((tid & 31) == 0) red[w] = s;
    __syncthreads();
    float rowsum = 0.0f;
#pragma unroll
    for (int i = 0; i < 8; ++i) rowsum += red[i];
    float lse = rowmax + __logf(rowsum);
    for (int c = tid; c < C; c += 256) p[c] = p[c] - lse;
}

// ---------------------------------------------------------------------------
// per-row NLL + argmax error, accumulated into out[0]=loss, out[1]=err
// ---------------------------------------------------------------------------
__global__ __launch_bounds__(256) void loss_err(const float* __restrict__ pout,
                                                const int* __restrict__ lab,
                                                float* out01, int C, int rows) {
    const float* p = pout + (size_t)blockIdx.x * C;
    const int tid = threadIdx.x;
    float bm = -__builtin_inff(); int bi = C;
    for (int c = tid; c < C; c += 256) {
        float v = p[c];
        if (v > bm || (v == bm && c < bi)) { bm = v; bi = c; }
    }
    __shared__ float vs[256]; __shared__ int is_[256];
    vs[tid] = bm; is_[tid] = bi;
    __syncthreads();
    for (int s = 128; s >= 1; s >>= 1) {
        if (tid < s) {
            float v2 = vs[tid + s]; int i2 = is_[tid + s];
            if (v2 > vs[tid] || (v2 == vs[tid] && i2 < is_[tid])) { vs[tid] = v2; is_[tid] = i2; }
        }
        __syncthreads();
    }
    if (tid == 0) {
        int l = lab[blockIdx.x];
        atomicAdd(&out01[0], -p[l] * (1.0f / rows));
        atomicAdd(&out01[1], (is_[0] != l) ? (1.0f / rows) : 0.0f);
    }
}

// ---------------------------------------------------------------------------
extern "C" void kernel_launch(void* const* d_in, const int* in_sizes, int n_in,
                              void* d_out, int out_size, void* d_ws, size_t ws_size,
                              hipStream_t stream) {
    (void)in_sizes; (void)n_in; (void)out_size; (void)ws_size;
    const float* x     = (const float*)d_in[0];
    const int*   lab   = (const int*)d_in[1];
    const float* wx0_w = (const float*)d_in[3];
    const float* wx0_b = (const float*)d_in[4];
    const float* wx_w  = (const float*)d_in[5];
    const float* wx_b  = (const float*)d_in[6];
    const float* uh_w  = (const float*)d_in[7];
    const float* ln_g  = (const float*)d_in[8];
    const float* ln_b  = (const float*)d_in[9];
    const float* fco_w = (const float*)d_in[10];
    const float* fco_b = (const float*)d_in[11];
    float* out = (float*)d_out;
    float* pout = out + 2;   // [TB, C_OUT] lives directly in d_out

    // workspace carve-out
    char* ws = (char*)d_ws; size_t off = 0;
    auto carve = [&](size_t bytes) { void* p = ws + off; off += (bytes + 255) & ~(size_t)255; return p; };
    _Float16* xh   = (_Float16*)carve((size_t)TB * D_PAD * 2);
    _Float16* w0h  = (_Float16*)carve((size_t)H_DIM * D_PAD * 2);
    _Float16* wxh  = (_Float16*)carve((size_t)3 * H_DIM * H_DIM * 2);
    _Float16* uhh  = (_Float16*)carve((size_t)4 * H_DIM * H_DIM * 2);
    _Float16* fcoh = (_Float16*)carve((size_t)C_PAD * H_DIM * 2);
    float*    wxo  = (float*)carve((size_t)TB * H_DIM * 4);
    _Float16* hh   = (_Float16*)carve((size_t)TB * H_DIM * 2);
    float*    preb = (float*)carve((size_t)TB * H_DIM * 4);

    zero2<<<1, 32, 0, stream>>>(out);

    auto cvt = [&](const float* s, _Float16* d, int rows, int K, int rowsPad, int Kpad) {
        long total = (long)rowsPad * Kpad;
        cvt_pad_f16<<<(unsigned)((total + 255) / 256), 256, 0, stream>>>(s, d, rows, K, rowsPad, Kpad);
    };
    cvt(x,     xh,   TB,        D_IN,  TB,        D_PAD);
    cvt(wx0_w, w0h,  H_DIM,     D_IN,  H_DIM,     D_PAD);
    cvt(wx_w,  wxh,  3 * H_DIM, H_DIM, 3 * H_DIM, H_DIM);
    cvt(uh_w,  uhh,  4 * H_DIM, H_DIM, 4 * H_DIM, H_DIM);
    cvt(fco_w, fcoh, C_OUT,     H_DIM, C_PAD,     H_DIM);

    dim3 gH(H_DIM / 128, TB / 128);                 // (8, 100)

    // layer 0
    gemm_wmma_f16<<<gH, 256, 0, stream>>>(xh, w0h, wx0_b, wxo, D_PAD, H_DIM, H_DIM);
    rnn_recurrence<<<1, 1024, REC_LDS_BYTES, stream>>>(wxo, preb, uhh, ln_g, ln_b, hh, 1);
    // layers 1..3
    for (int i = 1; i < 4; ++i) {
        gemm_wmma_f16<<<gH, 256, 0, stream>>>(hh, wxh + (size_t)(i - 1) * H_DIM * H_DIM,
                                              wx_b + (size_t)(i - 1) * H_DIM, wxo,
                                              H_DIM, H_DIM, H_DIM);
        rnn_recurrence<<<1, 1024, REC_LDS_BYTES, stream>>>(wxo, preb,
                                                    uhh + (size_t)i * H_DIM * H_DIM,
                                                    ln_g + (size_t)i * H_DIM,
                                                    ln_b + (size_t)i * H_DIM, hh, 0);
    }
    // output projection -> logits directly in d_out, then in-place log_softmax
    dim3 gC(C_PAD / 128, TB / 128);                 // (28, 100)
    gemm_wmma_f16<<<gC, 256, 0, stream>>>(hh, fcoh, fco_b, pout, H_DIM, C_OUT, C_OUT);
    log_softmax_inplace<<<TB, 256, 0, stream>>>(pout, C_OUT);
    loss_err<<<TB, 256, 0, stream>>>(pout, lab, out, C_OUT, TB);
}
